// BatchedGAT4BGRL_47820165874091
// MI455X (gfx1250) — compile-verified, matrix operand order checked by reading, not run on hardware
//
#include <hip/hip_runtime.h>
#include <hip/hip_bf16.h>
#include <cstdint>
#include <cstddef>

// ---------------------------------------------------------------------------
// Types for WMMA fragments (gfx1250 / CDNA5, wave32)
// ---------------------------------------------------------------------------
typedef __attribute__((ext_vector_type(16))) __bf16 v16bf;
typedef __attribute__((ext_vector_type(8)))  __bf16 v8bf;
typedef __attribute__((ext_vector_type(8)))  float  v8f;
typedef __attribute__((ext_vector_type(4)))  unsigned int u32x4;
typedef __attribute__((ext_vector_type(8)))  int i32x8;
typedef __attribute__((ext_vector_type(4)))  int i32x4;

__device__ __forceinline__ unsigned short f32_to_bf16(float f) {
  unsigned int u = __float_as_uint(f);
  u += 0x7FFFu + ((u >> 16) & 1u);           // round-to-nearest-even
  return (unsigned short)(u >> 16);
}
__device__ __forceinline__ float bf16_to_f32(unsigned short h) {
  return __uint_as_float(((unsigned int)h) << 16);
}

// ---------------------------------------------------------------------------
// Tensor Data Mover support (gfx1250). Two probed toolchains differ in builtin
// arity: ROCm 7.2 / clang-22 = 5 args, amdgpu-toolchain / clang-23 = 6 args.
// ---------------------------------------------------------------------------
#if defined(__AMDGCN__) && __has_builtin(__builtin_amdgcn_tensor_load_to_lds) && \
    __has_builtin(__builtin_amdgcn_s_wait_tensorcnt)
#define HAVE_TDM 1
#endif

#ifdef HAVE_TDM
// D# group1 for a 2D tile: tile 32 (K) x 128 (rows) of 2-byte elements out of a
// rows x K row-major tensor; LDS padding: +4 DWORDs after every 16 DWORDs
// (64B row -> 80B LDS row stride, de-conflicts ds_load_b128 fragment reads).
__device__ __forceinline__ i32x8 tdm_group1(int K, int rows) {
  i32x8 g;
  g[0] = 0x00010000 |        // data_size = 1 (2 bytes), workgroup_mask = 0
         (1 << 20)  |        // pad_enable
         (3 << 22)  |        // pad_interval: 16 DWORDs
         (3 << 25);          // pad_amount:   4 DWORDs
  g[1] = (K & 0xFFFF) << 16;                         // tensor_dim0 [15:0]
  g[2] = ((K >> 16) & 0xFFFF) | ((rows & 0xFFFF) << 16);  // dim0 hi | dim1 lo
  g[3] = ((rows >> 16) & 0xFFFF) | (32 << 16);       // dim1 hi | tile_dim0 = 32
  g[4] = 128;                                        // tile_dim1 = 128 rows
  g[5] = K;                                          // tensor_dim0_stride
  g[6] = 0;
  g[7] = 0;
  return g;
}

__device__ __forceinline__ void tdm_load_2d(unsigned long long gaddr,
                                            unsigned int lds_off, i32x8 g1) {
  u32x4 g0;
  g0[0] = 1u;                                        // count = 1 (valid user D#)
  g0[1] = lds_off;                                   // lds_addr
  g0[2] = (unsigned int)gaddr;                       // global_addr[31:0]
  g0[3] = ((unsigned int)(gaddr >> 32) & 0x01FFFFFFu) | 0x80000000u; // hi | type=2
  i32x4 z4 = {};
#if defined(__clang_major__) && (__clang_major__ >= 23)
  i32x8 z8 = {};
  __builtin_amdgcn_tensor_load_to_lds(g0, g1, z4, z4, z8, 0);
#else
  __builtin_amdgcn_tensor_load_to_lds(g0, g1, z4, z4, 0);
#endif
}
#endif  // HAVE_TDM

// ---------------------------------------------------------------------------
// C = A (MxK, bf16 row-major) @ B^T (B is NxK bf16 row-major), f32 accumulate.
// Block: 256 threads = 8 waves; block tile 128x128; wave tile 64x32 (4x2 WMMA).
// TDM path: double-buffered 128x32 A/B tiles staged in LDS by the tensor DMA,
// consumed via ds_load_b128; TENSORcnt + barriers for pipelining.
// All M,N multiples of 128 and K multiples of 32 for this workload.
// ---------------------------------------------------------------------------
#define TDM_ROW_HALVES 40         /* 80-byte padded LDS row stride */
#define TDM_TILE_BYTES 10240      /* 128 rows * 80B */
#define TDM_BUF_BYTES  20480      /* A tile + B tile */
#define TDM_LDS_BYTES  40960      /* double buffered */

__global__ __launch_bounds__(256) void gemm_bt_bf16(
    const unsigned short* __restrict__ A,
    const unsigned short* __restrict__ B,
    float* __restrict__ Cf,            // f32 output (or nullptr)
    unsigned short* __restrict__ Cb,   // bf16 output (or nullptr)
    int M, int N, int K)
{
  const int lane = threadIdx.x & 31;
  const int wave = threadIdx.x >> 5;
  const int wm = wave >> 2;            // 0..1
  const int wn = wave & 3;             // 0..3
  const int lr = lane & 15;            // row/col within 16x16 tile
  const int hi = lane >> 4;            // half-wave id
  const int kb = hi * 8;               // per-lane K sub-offset (halves)

  const int bm = blockIdx.x * 128 + wm * 64;
  const int bn = blockIdx.y * 128 + wn * 32;

  v8f zero = {};
  v8f acc[4][2];
#pragma unroll
  for (int i = 0; i < 4; ++i)
#pragma unroll
    for (int j = 0; j < 2; ++j) acc[i][j] = zero;

#ifdef HAVE_TDM
  extern __shared__ unsigned short smem[];   // dynamic LDS starts at offset 0
  const i32x8 g1A = tdm_group1(K, M);
  const i32x8 g1B = tdm_group1(K, N);
  const unsigned long long gA =
      (unsigned long long)(uintptr_t)A + ((size_t)blockIdx.x * 128) * (size_t)K * 2;
  const unsigned long long gB =
      (unsigned long long)(uintptr_t)B + ((size_t)blockIdx.y * 128) * (size_t)K * 2;

  if (wave == 0) {                     // prologue: fill buffer 0
    tdm_load_2d(gA, 0u, g1A);
    tdm_load_2d(gB, TDM_TILE_BYTES, g1B);
  }
  int p = 0;
  for (int kk = 0; kk < K; kk += 32) {
    const bool more = (kk + 32) < K;
    if (wave == 0) {
      if (more) {                      // stream next K-slab into the other buffer
        const unsigned int nb = (unsigned)(p ^ 1) * TDM_BUF_BYTES;
        tdm_load_2d(gA + (size_t)(kk + 32) * 2, nb, g1A);
        tdm_load_2d(gB + (size_t)(kk + 32) * 2, nb + TDM_TILE_BYTES, g1B);
        __builtin_amdgcn_s_wait_tensorcnt((short)2);   // current pair landed
      } else {
        __builtin_amdgcn_s_wait_tensorcnt((short)0);
      }
    }
    __syncthreads();                   // tile visible to all waves

    const unsigned short* sA = smem + (size_t)p * (TDM_BUF_BYTES / 2);
    const unsigned short* sB = sA + (TDM_TILE_BYTES / 2);

    v16bf af[4], bf_[2];
#pragma unroll
    for (int i = 0; i < 4; ++i) {
      const unsigned short* ap = sA + (wm * 64 + i * 16 + lr) * TDM_ROW_HALVES + kb;
      v8bf lo = *(const v8bf*)ap;          // K = kb .. kb+7
      v8bf hh = *(const v8bf*)(ap + 16);   // K = kb+16 .. kb+23
      af[i] = __builtin_shufflevector(lo, hh, 0,1,2,3,4,5,6,7,8,9,10,11,12,13,14,15);
    }
#pragma unroll
    for (int j = 0; j < 2; ++j) {
      const unsigned short* bp = sB + (wn * 32 + j * 16 + lr) * TDM_ROW_HALVES + kb;
      v8bf lo = *(const v8bf*)bp;
      v8bf hh = *(const v8bf*)(bp + 16);
      bf_[j] = __builtin_shufflevector(lo, hh, 0,1,2,3,4,5,6,7,8,9,10,11,12,13,14,15);
    }
#pragma unroll
    for (int i = 0; i < 4; ++i)
#pragma unroll
      for (int j = 0; j < 2; ++j)
        acc[i][j] = __builtin_amdgcn_wmma_f32_16x16x32_bf16(
            false, af[i], false, bf_[j], (short)0, acc[i][j], false, false);

    __syncthreads();                   // done reading buf p before it is refilled
    p ^= 1;
  }
#else  // !HAVE_TDM: direct global-load fallback (also used by the host pass)
  const size_t Ks = (size_t)K;
  for (int kk = 0; kk < K; kk += 32) {
    v16bf af[4], bf_[2];
#pragma unroll
    for (int i = 0; i < 4; ++i) {
      const unsigned short* ap = A + (size_t)(bm + i * 16 + lr) * Ks + kk + kb;
      v8bf lo = *(const v8bf*)ap;
      v8bf hh = *(const v8bf*)(ap + 16);
      af[i] = __builtin_shufflevector(lo, hh, 0,1,2,3,4,5,6,7,8,9,10,11,12,13,14,15);
    }
#pragma unroll
    for (int j = 0; j < 2; ++j) {
      const unsigned short* bp = B + (size_t)(bn + j * 16 + lr) * Ks + kk + kb;
      v8bf lo = *(const v8bf*)bp;
      v8bf hh = *(const v8bf*)(bp + 16);
      bf_[j] = __builtin_shufflevector(lo, hh, 0,1,2,3,4,5,6,7,8,9,10,11,12,13,14,15);
    }
    if (kk + 32 < K) {
      __builtin_prefetch((const void*)(A + (size_t)(bm + lr) * Ks + kk + 32 + kb), 0, 3);
      __builtin_prefetch((const void*)(B + (size_t)(bn + lr) * Ks + kk + 32 + kb), 0, 3);
    }
#pragma unroll
    for (int i = 0; i < 4; ++i)
#pragma unroll
      for (int j = 0; j < 2; ++j)
        acc[i][j] = __builtin_amdgcn_wmma_f32_16x16x32_bf16(
            false, af[i], false, bf_[j], (short)0, acc[i][j], false, false);
  }
#endif

  // C/D layout: element r of lane L -> M = r + 8*(L>=16), N = L%16
#pragma unroll
  for (int i = 0; i < 4; ++i) {
#pragma unroll
    for (int j = 0; j < 2; ++j) {
#pragma unroll
      for (int r = 0; r < 8; ++r) {
        const int row = bm + i * 16 + hi * 8 + r;
        const int col = bn + j * 16 + lr;
        const float v = acc[i][j][r];
        if (Cf) Cf[(size_t)row * N + col] = v;
        if (Cb) Cb[(size_t)row * N + col] = f32_to_bf16(v);
      }
    }
  }
}

// ---------------------------------------------------------------------------
// Elementwise / small helper kernels
// ---------------------------------------------------------------------------
__global__ void cvt_f32_bf16(const float* __restrict__ in, unsigned short* __restrict__ out, size_t n) {
  size_t i = (size_t)blockIdx.x * blockDim.x + threadIdx.x;
  if (i < n) out[i] = f32_to_bf16(in[i]);
}

__global__ void fill_f32(float* __restrict__ p, float v, size_t n) {
  size_t i = (size_t)blockIdx.x * blockDim.x + threadIdx.x;
  if (i < n) p[i] = v;
}

// u[h*K+k] = sum_c att[h*256+c] * w[(h*256+c)*K + k]   (H=4, C=256 fixed)
__global__ void fold_att(const float* __restrict__ w, const float* __restrict__ att,
                         float* __restrict__ u, int K) {
  int idx = blockIdx.x * blockDim.x + threadIdx.x;
  if (idx >= 4 * K) return;
  int h = idx / K, k = idx - h * K;
  const float* wr = w + (size_t)h * 256 * K + k;
  const float* ar = att + h * 256;
  float s = 0.f;
  for (int c = 0; c < 256; ++c) s = fmaf(ar[c], wr[(size_t)c * K], s);
  u[idx] = s;
}

// al[n*4+h] = x[n,:] . u[h,:]   — one wave32 per node
__global__ __launch_bounds__(256) void al_kernel(const float* __restrict__ x,
                                                 const float* __restrict__ u,
                                                 float* __restrict__ al, int N, int K) {
  int gw = (int)(((size_t)blockIdx.x * blockDim.x + threadIdx.x) >> 5);
  int lane = threadIdx.x & 31;
  if (gw >= N) return;
  const float* xr = x + (size_t)gw * K;
  float s0 = 0.f, s1 = 0.f, s2 = 0.f, s3 = 0.f;
  for (int k = lane; k < K; k += 32) {
    float xv = xr[k];
    s0 = fmaf(xv, u[k], s0);
    s1 = fmaf(xv, u[K + k], s1);
    s2 = fmaf(xv, u[2 * K + k], s2);
    s3 = fmaf(xv, u[3 * K + k], s3);
  }
  for (int off = 16; off > 0; off >>= 1) {
    s0 += __shfl_down(s0, off, 32);
    s1 += __shfl_down(s1, off, 32);
    s2 += __shfl_down(s2, off, 32);
    s3 += __shfl_down(s3, off, 32);
  }
  if (lane == 0) {
    float* r = al + (size_t)gw * 4;
    r[0] = s0; r[1] = s1; r[2] = s2; r[3] = s3;
  }
}

// ---------------------------------------------------------------------------
// Edge-softmax passes (edges E plus `size` implicit self-loops)
// ---------------------------------------------------------------------------
__device__ __forceinline__ void edge_sd(int e, const int* src, const int* dst, int E,
                                        int& s, int& d) {
  if (e < E) { s = src[e]; d = dst[e]; } else { s = e - E; d = e - E; }
}

__device__ __forceinline__ float lrelu02(float l) { return l < 0.f ? 0.2f * l : l; }

__global__ void edge_max_k(const int* __restrict__ src, const int* __restrict__ dst,
                           int E, int size,
                           const float* __restrict__ als, const float* __restrict__ ald,
                           float* __restrict__ m) {
  int idx = blockIdx.x * blockDim.x + threadIdx.x;
  int tot = (E + size) * 4;
  if (idx >= tot) return;
  int e = idx >> 2, h = idx & 3;
  int s, d; edge_sd(e, src, dst, E, s, d);
  float l = lrelu02(als[(size_t)s * 4 + h] + ald[(size_t)d * 4 + h]);
  float* addr = &m[(size_t)d * 4 + h];
  // order-preserving float atomic-max (sign-split)
  if (l >= 0.f) atomicMax((int*)addr, __float_as_int(l));
  else          atomicMin((unsigned int*)addr, __float_as_uint(l));
}

__global__ void edge_sum_k(const int* __restrict__ src, const int* __restrict__ dst,
                           int E, int size,
                           const float* __restrict__ als, const float* __restrict__ ald,
                           const float* __restrict__ m, float* __restrict__ den) {
  int idx = blockIdx.x * blockDim.x + threadIdx.x;
  int tot = (E + size) * 4;
  if (idx >= tot) return;
  int e = idx >> 2, h = idx & 3;
  int s, d; edge_sd(e, src, dst, E, s, d);
  float l = lrelu02(als[(size_t)s * 4 + h] + ald[(size_t)d * 4 + h]);
  float ex = __expf(l - m[(size_t)d * 4 + h]);
  atomicAdd(&den[(size_t)d * 4 + h], ex);
}

// out[d, f] += alpha[h(f)] * h_src[s, f]  — one block per edge, hc = 1024
__global__ __launch_bounds__(256) void edge_scatter_k(
    const int* __restrict__ src, const int* __restrict__ dst, int E, int size,
    const float* __restrict__ als, const float* __restrict__ ald,
    const float* __restrict__ m, const float* __restrict__ den,
    const unsigned short* __restrict__ hb, float* __restrict__ out) {
  int e = blockIdx.x;
  int s, d; edge_sd(e, src, dst, E, s, d);
  float alpha[4];
#pragma unroll
  for (int h = 0; h < 4; ++h) {
    float l = lrelu02(als[(size_t)s * 4 + h] + ald[(size_t)d * 4 + h]);
    float ex = __expf(l - m[(size_t)d * 4 + h]);
    alpha[h] = ex / (den[(size_t)d * 4 + h] + 1e-16f);
  }
  const unsigned short* hr = hb + (size_t)s * 1024;
  float* orow = out + (size_t)d * 1024;
  for (int f = threadIdx.x; f < 1024; f += 256)
    atomicAdd(&orow[f], alpha[f >> 8] * bf16_to_f32(hr[f]));
}

// L3 only: red[n,c] = mean over 4 heads of out[n, h*256 + c]
__global__ void head_mean(const float* __restrict__ out, float* __restrict__ red, int size) {
  int idx = blockIdx.x * blockDim.x + threadIdx.x;
  if (idx >= size * 256) return;
  int n = idx >> 8, c = idx & 255;
  const float* r = out + (size_t)n * 1024;
  red[idx] = 0.25f * (r[c] + r[256 + c] + r[512 + c] + r[768 + c]);
}

__global__ void add_bias_skip(float* __restrict__ h, const float* __restrict__ sk,
                              const float* __restrict__ bc, const float* __restrict__ bs,
                              size_t total, int cols) {
  size_t idx = (size_t)blockIdx.x * blockDim.x + threadIdx.x;
  if (idx >= total) return;
  int j = (int)(idx % cols);
  h[idx] += sk[idx] + bc[j] + bs[j];
}

// Two-stage column stats (sum / sumsq accumulated with atomics, finalized below)
__global__ void colstat_partial(const float* __restrict__ h, float* __restrict__ sum,
                                float* __restrict__ ssq, int rows, int cols) {
  int col = blockIdx.x * blockDim.x + threadIdx.x;
  if (col >= cols) return;
  int nch = gridDim.y;
  int chunk = (rows + nch - 1) / nch;
  int r0 = blockIdx.y * chunk;
  int r1 = min(r0 + chunk, rows);
  float s = 0.f, q = 0.f;
  for (int r = r0; r < r1; ++r) {
    float v = h[(size_t)r * cols + col];
    s += v; q += v * v;
  }
  atomicAdd(&sum[col], s);
  atomicAdd(&ssq[col], q);
}

__global__ void colstat_final(float* __restrict__ mu, float* __restrict__ var,
                              int rows, int cols) {
  int j = blockIdx.x * blockDim.x + threadIdx.x;
  if (j >= cols) return;
  float m = mu[j] / (float)rows;
  float q = var[j] / (float)rows - m * m;
  mu[j] = m; var[j] = q;
}

__global__ void bn_elu(const float* __restrict__ h, const float* __restrict__ mu,
                       const float* __restrict__ var, const float* __restrict__ g,
                       const float* __restrict__ be, float* __restrict__ xout,
                       size_t total, int cols) {
  size_t idx = (size_t)blockIdx.x * blockDim.x + threadIdx.x;
  if (idx >= total) return;
  int j = (int)(idx % cols);
  float v = (h[idx] - mu[j]) * rsqrtf(var[j] + 1e-5f) * g[j] + be[j];
  xout[idx] = v > 0.f ? v : (__expf(v) - 1.f);
}

// ---------------------------------------------------------------------------
// Host orchestration
// ---------------------------------------------------------------------------
extern "C" void kernel_launch(void* const* d_in, const int* in_sizes, int n_in,
                              void* d_out, int out_size, void* d_ws, size_t ws_size,
                              hipStream_t stream) {
  (void)in_sizes; (void)n_in; (void)out_size; (void)ws_size;

  constexpr int N0 = 262144, S1 = 32768, S2 = 4096, S3 = 512, IN = 128;
  constexpr int E1 = 327680, E2 = 40960, E3 = 5120;

  const float* x0 = (const float*)d_in[0];

  struct Layer {
    const float *ws, *wd, *as_, *ad, *bc, *wsk, *bsk, *g, *be;
    const int *es, *ed;
    int N, size, din, dskip, E;
  } L[3];
  for (int i = 0; i < 3; ++i) {
    int b = 1 + 9 * i;
    L[i].ws  = (const float*)d_in[b + 0];
    L[i].wd  = (const float*)d_in[b + 1];
    L[i].as_ = (const float*)d_in[b + 2];
    L[i].ad  = (const float*)d_in[b + 3];
    L[i].bc  = (const float*)d_in[b + 4];
    L[i].wsk = (const float*)d_in[b + 5];
    L[i].bsk = (const float*)d_in[b + 6];
    L[i].g   = (const float*)d_in[b + 7];
    L[i].be  = (const float*)d_in[b + 8];
  }
  L[0].es = (const int*)d_in[28]; L[0].ed = (const int*)d_in[29];
  L[1].es = (const int*)d_in[30]; L[1].ed = (const int*)d_in[31];
  L[2].es = (const int*)d_in[32]; L[2].ed = (const int*)d_in[33];
  L[0].N = N0; L[0].size = S1; L[0].din = IN;   L[0].dskip = 1024; L[0].E = E1;
  L[1].N = S1; L[1].size = S2; L[1].din = 1024; L[1].dskip = 1024; L[1].E = E2;
  L[2].N = S2; L[2].size = S3; L[2].din = 1024; L[2].dskip = 256;  L[2].E = E3;

  // Workspace carve (max sizes across layers)
  char* base = (char*)d_ws;
  size_t off = 0;
  auto carve = [&](size_t bytes) -> void* {
    void* r = base + off;
    off += (bytes + 255) & ~(size_t)255;
    return r;
  };
  unsigned short* XB   = (unsigned short*)carve((size_t)33554432 * 2);   // bf16 x
  float*          XF   = (float*)carve((size_t)33554432 * 4);            // f32 next-layer x
  unsigned short* HB   = (unsigned short*)carve((size_t)268435456 * 2);  // bf16 h_src
  unsigned short* WSB  = (unsigned short*)carve((size_t)1048576 * 2);    // bf16 w_src
  unsigned short* WSKB = (unsigned short*)carve((size_t)1048576 * 2);    // bf16 w_skip
  float* U    = (float*)carve((size_t)4 * 1024 * 4);
  float* V    = (float*)carve((size_t)4 * 1024 * 4);
  float* ALS  = (float*)carve((size_t)262144 * 4 * 4);
  float* ALD  = (float*)carve((size_t)32768 * 4 * 4);
  float* MM   = (float*)carve((size_t)32768 * 4 * 4);
  float* DEN  = (float*)carve((size_t)32768 * 4 * 4);
  float* OUTB = (float*)carve((size_t)32768 * 1024 * 4);
  float* SKB  = (float*)carve((size_t)32768 * 1024 * 4);
  float* RED  = (float*)carve((size_t)512 * 256 * 4);
  float* MUB  = (float*)carve((size_t)1024 * 4);
  float* VARB = (float*)carve((size_t)1024 * 4);

  const float* xf = x0;
  for (int li = 0; li < 3; ++li) {
    const Layer& Y = L[li];
    const int N = Y.N, sz = Y.size, K = Y.din, E = Y.E, dskip = Y.dskip;

    // x -> bf16
    {
      size_t n = (size_t)N * K;
      cvt_f32_bf16<<<(unsigned)((n + 255) / 256), 256, 0, stream>>>(xf, XB, n);
    }
    // fold attention vectors: al_* becomes a plain x@u^T
    fold_att<<<(4 * K + 255) / 256, 256, 0, stream>>>(Y.ws, Y.as_, U, K);
    fold_att<<<(4 * K + 255) / 256, 256, 0, stream>>>(Y.wd, Y.ad, V, K);
    al_kernel<<<(N * 32 + 255) / 256, 256, 0, stream>>>(xf, U, ALS, N, K);
    al_kernel<<<(sz * 32 + 255) / 256, 256, 0, stream>>>(xf, V, ALD, sz, K);

    // h_src = x @ w_src^T  (WMMA bf16 via TDM-staged LDS tiles)
    {
      size_t n = (size_t)1024 * K;
      cvt_f32_bf16<<<(unsigned)((n + 255) / 256), 256, 0, stream>>>(Y.ws, WSB, n);
      dim3 g(N / 128, 1024 / 128);
      gemm_bt_bf16<<<g, 256, TDM_LDS_BYTES, stream>>>(XB, WSB, nullptr, HB, N, 1024, K);
    }

    // edge softmax + scatter
    fill_f32<<<(sz * 4 + 255) / 256, 256, 0, stream>>>(MM, -3.0e38f, (size_t)sz * 4);
    hipMemsetAsync(DEN, 0, (size_t)sz * 4 * sizeof(float), stream);
    hipMemsetAsync(OUTB, 0, (size_t)sz * 1024 * sizeof(float), stream);
    int Etot = E + sz;
    edge_max_k<<<(unsigned)(((size_t)Etot * 4 + 255) / 256), 256, 0, stream>>>(
        Y.es, Y.ed, E, sz, ALS, ALD, MM);
    edge_sum_k<<<(unsigned)(((size_t)Etot * 4 + 255) / 256), 256, 0, stream>>>(
        Y.es, Y.ed, E, sz, ALS, ALD, MM, DEN);
    edge_scatter_k<<<Etot, 256, 0, stream>>>(Y.es, Y.ed, E, sz, ALS, ALD, MM, DEN, HB, OUTB);

    // skip = x[:size] @ w_skip^T  (WMMA bf16, f32 output)
    {
      size_t n = (size_t)dskip * K;
      cvt_f32_bf16<<<(unsigned)((n + 255) / 256), 256, 0, stream>>>(Y.wsk, WSKB, n);
      dim3 g(sz / 128, dskip / 128);
      gemm_bt_bf16<<<g, 256, TDM_LDS_BYTES, stream>>>(XB, WSKB, SKB, nullptr, sz, dskip, K);
    }

    float* hpre = OUTB;
    int cols = dskip;
    if (li == 2) {   // concat=False -> head mean
      head_mean<<<(sz * 256 + 255) / 256, 256, 0, stream>>>(OUTB, RED, sz);
      hpre = RED;
    }
    size_t total = (size_t)sz * cols;
    add_bias_skip<<<(unsigned)((total + 255) / 256), 256, 0, stream>>>(
        hpre, SKB, Y.bc, Y.bsk, total, cols);

    hipMemsetAsync(MUB, 0, cols * sizeof(float), stream);
    hipMemsetAsync(VARB, 0, cols * sizeof(float), stream);
    colstat_partial<<<dim3(cols / 256, 32), 256, 0, stream>>>(hpre, MUB, VARB, sz, cols);
    colstat_final<<<(cols + 255) / 256, 256, 0, stream>>>(MUB, VARB, sz, cols);

    float* xout = (li == 2) ? (float*)d_out : XF;
    bn_elu<<<(unsigned)((total + 255) / 256), 256, 0, stream>>>(
        hpre, MUB, VARB, Y.g, Y.be, xout, total, cols);
    xf = XF;
  }
}
